// CrossAttention_71150428225550
// MI455X (gfx1250) — matrix operand order repaired and checked by reading, not compiled, not run on hardware
//
#include <hip/hip_runtime.h>
#include <hip/hip_bf16.h>

typedef __attribute__((ext_vector_type(16))) __bf16 v16bf;
typedef __attribute__((ext_vector_type(8)))  __bf16 v8bf;
typedef __attribute__((ext_vector_type(8)))  float  v8f;

#define DMODEL 512
#define NHEADS 8
#define DHEAD  64

// 1/sqrt(Dh) * log2(e): folded into the Q projection so the flash kernel can
// use exp2 directly with unscaled scores.
#define QSCALE 0.18033688011112042f

__device__ __forceinline__ __bf16 f2bf(float f) {
  unsigned u = __builtin_bit_cast(unsigned, f);
  u += 0x7FFFu + ((u >> 16) & 1u);            // round-to-nearest-even
  unsigned short h = (unsigned short)(u >> 16);
  return __builtin_bit_cast(__bf16, h);
}
__device__ __forceinline__ __bf16 f2bf_rz(float f) {   // truncate: 1 VALU op
  unsigned u = __builtin_bit_cast(unsigned, f);
  unsigned short h = (unsigned short)(u >> 16);
  return __builtin_bit_cast(__bf16, h);
}
__device__ __forceinline__ float fmaxx(float a, float b) {  // no NaN canonicalize
  return a > b ? a : b;
}
__device__ __forceinline__ float exp2x(float x) {           // v_exp_f32
  return __builtin_amdgcn_exp2f(x);
}

__device__ __forceinline__ v8f wmma_bf16(v16bf a, v16bf b, v8f c) {
  // D(16x16,f32) = A(16x32,bf16) * B(32x16,bf16) + C
  return __builtin_amdgcn_wmma_f32_16x16x32_bf16(false, a, false, b, (short)0, c,
                                                 false, false);
}

// ---- 16-lane butterfly reductions in the VALU pipe via DPP (no LDS) --------
template<int CTRL>
__device__ __forceinline__ float dpp_mov(float v) {
  return __builtin_bit_cast(float,
      __builtin_amdgcn_update_dpp(0, __builtin_bit_cast(int, v), CTRL, 0xf, 0xf,
                                  true));
}
__device__ __forceinline__ float red_max16(float v) {
  v = fmaxx(v, dpp_mov<0xB1>(v));    // quad_perm [1,0,3,2]  (xor 1)
  v = fmaxx(v, dpp_mov<0x4E>(v));    // quad_perm [2,3,0,1]  (xor 2)
  v = fmaxx(v, dpp_mov<0x141>(v));   // row_half_mirror      (xor 4 after quads)
  v = fmaxx(v, dpp_mov<0x140>(v));   // row_mirror           (xor 8 after eighths)
  return v;
}
__device__ __forceinline__ float red_sum16(float v) {
  v += dpp_mov<0xB1>(v);
  v += dpp_mov<0x4E>(v);
  v += dpp_mov<0x141>(v);
  v += dpp_mov<0x140>(v);
  return v;
}

// ---------------------------------------------------------------------------
// One-pass f32 -> bf16 conversion (row-major), 8 elements/thread.
// ---------------------------------------------------------------------------
__global__ __launch_bounds__(256)
void cvt_f32_bf16(const float* __restrict__ in, __bf16* __restrict__ out, int n) {
  int i = (blockIdx.x * 256 + threadIdx.x) * 8;
  if (i + 8 <= n) {
    float4 a = *(const float4*)(in + i);
    float4 b = *(const float4*)(in + i + 4);
    v8bf o;
    o[0] = f2bf(a.x); o[1] = f2bf(a.y); o[2] = f2bf(a.z); o[3] = f2bf(a.w);
    o[4] = f2bf(b.x); o[5] = f2bf(b.y); o[6] = f2bf(b.z); o[7] = f2bf(b.w);
    *(v8bf*)(out + i) = o;
  }
}

// ---------------------------------------------------------------------------
// W (f32, [k][c] row-major 512x512) -> Wt (bf16, [c][k] column-major).
// ---------------------------------------------------------------------------
__global__ __launch_bounds__(256)
void transpose_w_bf16(const float* __restrict__ W, __bf16* __restrict__ Wt) {
  __shared__ float tile[32][33];
  const int tx = threadIdx.x & 31, ty = threadIdx.x >> 5;   // ty: 0..7
  const int kb = blockIdx.y * 32, cb = blockIdx.x * 32;
  #pragma unroll
  for (int i = 0; i < 4; ++i)
    tile[ty + 8 * i][tx] = W[(size_t)(kb + ty + 8 * i) * DMODEL + cb + tx];
  __syncthreads();
  #pragma unroll
  for (int i = 0; i < 4; ++i)
    Wt[(size_t)(cb + ty + 8 * i) * DMODEL + kb + tx] = f2bf(tile[tx][ty + 8 * i]);
}

// ---------------------------------------------------------------------------
// GEMM: out(nrows x 512) = (A(bf16) @ W + bias) * out_scale,
// W as bf16 W^T (column-major). Block 128 = 4 waves; wave = 16 rows x 64 cols.
// Grid (nrows/64, 8). No LDS.
// ---------------------------------------------------------------------------
template<bool OUT_F32, bool OUT_VT>
__global__ __launch_bounds__(128)
void gemm_bf16_d512(const __bf16* __restrict__ A, const __bf16* __restrict__ Wt,
                    const float* __restrict__ bias, void* __restrict__ outp,
                    int rows_per_batch, float out_scale)
{
  const int wave = threadIdx.x >> 5, lane = threadIdx.x & 31;
  const int ln = lane & 15, g = lane >> 4;
  const int row0 = blockIdx.x * 64 + wave * 16;
  const int col0 = blockIdx.y * 64;

  v8f acc[4] = {};
  const __bf16* ar = A + (size_t)(row0 + ln) * DMODEL;

  #pragma unroll 4
  for (int k0 = 0; k0 < DMODEL; k0 += 32) {
    v8bf a0 = *(const v8bf*)(ar + k0 + g * 8);
    v8bf a1 = *(const v8bf*)(ar + k0 + 16 + g * 8);
    v16bf a;
    #pragma unroll
    for (int e = 0; e < 8; ++e) { a[e] = a0[e]; a[8 + e] = a1[e]; }

    #pragma unroll
    for (int nt = 0; nt < 4; ++nt) {
      const __bf16* wp = Wt + (size_t)(col0 + nt * 16 + ln) * DMODEL + k0 + g * 8;
      v8bf b0 = *(const v8bf*)wp;
      v8bf b1 = *(const v8bf*)(wp + 16);
      v16bf bb;
      #pragma unroll
      for (int e = 0; e < 8; ++e) { bb[e] = b0[e]; bb[8 + e] = b1[e]; }
      acc[nt] = wmma_bf16(a, bb, acc[nt]);
    }
  }

  #pragma unroll
  for (int nt = 0; nt < 4; ++nt) {
    int c = col0 + nt * 16 + ln;
    float bz = bias[c];
    #pragma unroll
    for (int r = 0; r < 8; ++r) {
      int row = row0 + g * 8 + r;
      float val = (acc[nt][r] + bz) * out_scale;
      if constexpr (OUT_VT) {
        int b = row / rows_per_batch, mm = row % rows_per_batch;
        int h = c >> 6, d = c & 63;
        ((__bf16*)outp)[(((size_t)b * NHEADS + h) * DHEAD + d) * rows_per_batch + mm]
            = f2bf(val);
      } else if constexpr (OUT_F32) {
        ((float*)outp)[(size_t)row * DMODEL + c] = val;
      } else {
        ((__bf16*)outp)[(size_t)row * DMODEL + c] = f2bf(val);
      }
    }
  }
}

// ---------------------------------------------------------------------------
// Flash cross-attention. Block = 256 threads = 8 waves; each wave owns 16
// query rows of one (b,h); block covers 128 queries. K/V streamed in 64-key
// chunks, staged once per block into LDS with async loads (ASYNCcnt,
// double-buffered), shared by all 8 waves. Online softmax with DPP, in
// log2-domain (scale folded into Q projection) so exp == v_exp_f32 directly.
// Q,K: [b, t/m, h, d] bf16; V transposed [b][h][d][m] bf16. Z: [b,t,h,d] bf16.
// ---------------------------------------------------------------------------
#define KVP 72   // padded LDS row pitch (bf16 elems): conflict-free b128 reads

__global__ __launch_bounds__(256)
void flash_attn(const __bf16* __restrict__ Q, const __bf16* __restrict__ K,
                const __bf16* __restrict__ Vt, __bf16* __restrict__ Z,
                int T, int M)
{
  __shared__ __bf16 Ksh[2][64][KVP];   // [buf][key][d]
  __shared__ __bf16 Vsh[2][64][KVP];   // [buf][d][key]
  __shared__ __bf16 P[8][16][KVP];     // per-wave C->A shuffle buffer

  const int tid = threadIdx.x;
  const int wave = tid >> 5, lane = tid & 31;
  const int ln = lane & 15, g = lane >> 4;
  const int h = blockIdx.y, b = blockIdx.z;
  const int t0 = (blockIdx.x * 8 + wave) * 16;

  const __bf16* Qb = Q + ((size_t)b * T) * DMODEL + h * DHEAD;
  const __bf16* Kb = K + ((size_t)b * M) * DMODEL + h * DHEAD;
  const __bf16* Vb = Vt + ((size_t)(b * NHEADS + h) * DHEAD) * M;
  const unsigned long long kbase = (unsigned long long)(size_t)Kb;
  const unsigned long long vbase = (unsigned long long)(size_t)Vb;

  // async-stage one 64-key K tile (rows=key, cols=d) and V tile (rows=d,
  // cols=key): 64 rows x 128B, 16B per lane, 2 instructions per array/thread.
  auto issue = [&](int jj, int bb) {
    #pragma unroll
    for (int i = 0; i < 2; ++i) {
      int s = tid + i * 256;              // 0..511
      int row = s >> 3, seg = s & 7;      // row: key (K) / d (V), seg: 16B chunk
      unsigned klds = (unsigned)(size_t)&Ksh[bb][row][seg * 8];
      unsigned kofs = (unsigned)(((jj + row) * DMODEL + seg * 8) * 2);
      asm volatile("global_load_async_to_lds_b128 %0, %1, %2"
                   :: "v"(klds), "v"(kofs), "s"(kbase) : "memory");
      unsigned vlds = (unsigned)(size_t)&Vsh[bb][row][seg * 8];
      unsigned vofs = (unsigned)(((unsigned)row * (unsigned)M + jj + seg * 8) * 2);
      asm volatile("global_load_async_to_lds_b128 %0, %1, %2"
                   :: "v"(vlds), "v"(vofs), "s"(vbase) : "memory");
    }
  };

  // Q fragments (Dh = 64 -> two k-steps), loaded once from global
  v16bf qf[2];
  {
    const __bf16* qr = Qb + (size_t)(t0 + ln) * DMODEL;
    #pragma unroll
    for (int ks = 0; ks < 2; ++ks) {
      v8bf c0 = *(const v8bf*)(qr + ks * 32 + g * 8);
      v8bf c1 = *(const v8bf*)(qr + ks * 32 + 16 + g * 8);
      #pragma unroll
      for (int e = 0; e < 8; ++e) { qf[ks][e] = c0[e]; qf[ks][8 + e] = c1[e]; }
    }
  }

  float mrow[8], lrow[8];
  #pragma unroll
  for (int r = 0; r < 8; ++r) { mrow[r] = -3.0e38f; lrow[r] = 0.f; }
  v8f zacc[4] = {};

  issue(0, 0);
  int buf = 0;
  for (int j = 0; j < M; j += 64) {
    if (j + 64 < M) {
      issue(j + 64, buf ^ 1);                       // prefetch next chunk
      asm volatile("s_wait_asynccnt 4" ::: "memory");   // this chunk done
    } else {
      asm volatile("s_wait_asynccnt 0" ::: "memory");
    }
    __syncthreads();

    // S(16x64) = Q(16x64) . K^T : 4 n-tiles x 2 k-steps (K from LDS)
    v8f s[4] = {};
    #pragma unroll
    for (int c = 0; c < 4; ++c) {
      const __bf16* kr = &Ksh[buf][c * 16 + ln][0];
      #pragma unroll
      for (int ks = 0; ks < 2; ++ks) {
        v8bf b0 = *(const v8bf*)(kr + ks * 32 + g * 8);
        v8bf b1 = *(const v8bf*)(kr + ks * 32 + 16 + g * 8);
        v16bf kf;
        #pragma unroll
        for (int e = 0; e < 8; ++e) { kf[e] = b0[e]; kf[8 + e] = b1[e]; }
        s[c] = wmma_bf16(qf[ks], kf, s[c]);
      }
    }

    // online softmax (log2 domain); C-frag row = g*8+r lives in a 16-lane half
    #pragma unroll
    for (int r = 0; r < 8; ++r) {
      float mx = red_max16(fmaxx(fmaxx(s[0][r], s[1][r]),
                                 fmaxx(s[2][r], s[3][r])));
      float mnew = fmaxx(mrow[r], mx);
      float corr = exp2x(mrow[r] - mnew);
      float p0 = exp2x(s[0][r] - mnew), p1 = exp2x(s[1][r] - mnew);
      float p2 = exp2x(s[2][r] - mnew), p3 = exp2x(s[3][r] - mnew);
      float rs = red_sum16((p0 + p1) + (p2 + p3));
      lrow[r] = lrow[r] * corr + rs;
      mrow[r] = mnew;
      #pragma unroll
      for (int nt = 0; nt < 4; ++nt) zacc[nt][r] *= corr;
      P[wave][g * 8 + r][ln]      = f2bf_rz(p0);
      P[wave][g * 8 + r][16 + ln] = f2bf_rz(p1);
      P[wave][g * 8 + r][32 + ln] = f2bf_rz(p2);
      P[wave][g * 8 + r][48 + ln] = f2bf_rz(p3);
    }
    asm volatile("s_wait_dscnt 0" ::: "memory");

    // Z(16x64) += P(16x64) . V(64x64): 2 k-steps x 4 n-tiles (V from LDS)
    #pragma unroll
    for (int ks = 0; ks < 2; ++ks) {
      v16bf pf;
      {
        const __bf16* pr = &P[wave][ln][ks * 32];
        v8bf c0 = *(const v8bf*)(pr + g * 8);
        v8bf c1 = *(const v8bf*)(pr + 16 + g * 8);
        #pragma unroll
        for (int e = 0; e < 8; ++e) { pf[e] = c0[e]; pf[8 + e] = c1[e]; }
      }
      #pragma unroll
      for (int nt = 0; nt < 4; ++nt) {
        const __bf16* vr = &Vsh[buf][nt * 16 + ln][ks * 32 + g * 8];
        v8bf b0 = *(const v8bf*)vr;
        v8bf b1 = *(const v8bf*)(vr + 16);
        v16bf vf;
        #pragma unroll
        for (int e = 0; e < 8; ++e) { vf[e] = b0[e]; vf[8 + e] = b1[e]; }
        zacc[nt] = wmma_bf16(pf, vf, zacc[nt]);
      }
    }
    __syncthreads();    // everyone done with buf before it is re-filled
    buf ^= 1;
  }

  float rinv[8];
  #pragma unroll
  for (int r = 0; r < 8; ++r) rinv[r] = 1.0f / lrow[r];
  #pragma unroll
  for (int nt = 0; nt < 4; ++nt) {
    #pragma unroll
    for (int r = 0; r < 8; ++r) {
      int t = t0 + g * 8 + r;
      int d = nt * 16 + ln;
      Z[((size_t)(b * T + t)) * DMODEL + h * DHEAD + d] = f2bf(zacc[nt][r] * rinv[r]);
    }
  }
}

// ---------------------------------------------------------------------------
extern "C" void kernel_launch(void* const* d_in, const int* in_sizes, int n_in,
                              void* d_out, int out_size, void* d_ws, size_t ws_size,
                              hipStream_t stream) {
  (void)n_in; (void)out_size; (void)ws_size;
  const float* x   = (const float*)d_in[0];
  const float* mem = (const float*)d_in[1];
  const float* Wq  = (const float*)d_in[2];
  const float* bq  = (const float*)d_in[3];
  const float* Wk  = (const float*)d_in[4];
  const float* bk  = (const float*)d_in[5];
  const float* Wv  = (const float*)d_in[6];
  const float* bv  = (const float*)d_in[7];
  const float* Wo  = (const float*)d_in[8];
  const float* bo  = (const float*)d_in[9];

  const int B  = 4;
  const int BT = in_sizes[0] / DMODEL;   // B*T = 8192
  const int BM = in_sizes[1] / DMODEL;   // B*M = 16384
  const int T  = BT / B, M = BM / B;
  const size_t W2 = (size_t)DMODEL * DMODEL;

  // Workspace (bf16): xb(8MB, reused as Zb) | memb(16MB) | Wt x4 (2MB) |
  //                   Qb(8MB) | Kb(16MB) | Vt(16MB)  -> ~66MB total
  __bf16* xb   = (__bf16*)d_ws;                       // also Zb later
  __bf16* memb = xb + (size_t)BT * DMODEL;
  __bf16* Wqt  = memb + (size_t)BM * DMODEL;
  __bf16* Wkt  = Wqt + W2;
  __bf16* Wvt  = Wkt + W2;
  __bf16* Wot  = Wvt + W2;
  __bf16* Qb   = Wot + W2;
  __bf16* Kb   = Qb + (size_t)BT * DMODEL;
  __bf16* Vt   = Kb + (size_t)BM * DMODEL;
  __bf16* Zb   = xb;

  // 1) precision conversion (one pass, bandwidth bound)
  cvt_f32_bf16<<<dim3((BT * DMODEL) / (256 * 8)), 256, 0, stream>>>(x, xb, BT * DMODEL);
  cvt_f32_bf16<<<dim3((BM * DMODEL) / (256 * 8)), 256, 0, stream>>>(mem, memb, BM * DMODEL);
  transpose_w_bf16<<<dim3(16, 16), 256, 0, stream>>>(Wq, Wqt);
  transpose_w_bf16<<<dim3(16, 16), 256, 0, stream>>>(Wk, Wkt);
  transpose_w_bf16<<<dim3(16, 16), 256, 0, stream>>>(Wv, Wvt);
  transpose_w_bf16<<<dim3(16, 16), 256, 0, stream>>>(Wo, Wot);

  // 2) projections (all-WMMA, no LDS); Q carries 1/sqrt(Dh)*log2(e)
  dim3 blk(128);
  gemm_bf16_d512<false, false><<<dim3(BT / 64, 8), blk, 0, stream>>>(xb,   Wqt, bq, Qb, 1, QSCALE);
  gemm_bf16_d512<false, false><<<dim3(BM / 64, 8), blk, 0, stream>>>(memb, Wkt, bk, Kb, 1, 1.0f);
  gemm_bf16_d512<false, true ><<<dim3(BM / 64, 8), blk, 0, stream>>>(memb, Wvt, bv, Vt, M, 1.0f);

  // 3) flash attention (async LDS staging, 8 waves/block)
  flash_attn<<<dim3(T / 128, NHEADS, B), dim3(256), 0, stream>>>(Qb, Kb, Vt, Zb, T, M);

  // 4) output projection -> f32
  gemm_bf16_d512<true, false><<<dim3(BT / 64, 8), blk, 0, stream>>>(Zb, Wot, bo,
                                                                    (float*)d_out, 1, 1.0f);
}
